// BrepEncoder_10995116277858
// MI455X (gfx1250) — matrix operand order repaired and checked by reading, not compiled, not run on hardware
//
#include <hip/hip_runtime.h>

#define HID 256
#define NHEAD 4
#define HDIM 64
#define OUTD 512
#define ENC_NEG_INF 0x007FFFFFu  // ordered-uint encoding of -inf

typedef __attribute__((ext_vector_type(16))) _Float16 v16h;
typedef __attribute__((ext_vector_type(8)))  float    v8f;

// ---------- helpers ----------
__device__ __forceinline__ unsigned fenc(float f) {
  unsigned i = __float_as_uint(f);
  return (i & 0x80000000u) ? ~i : (i | 0x80000000u);
}
__device__ __forceinline__ float fdec(unsigned e) {
  unsigned i = (e & 0x80000000u) ? (e & 0x7FFFFFFFu) : ~e;
  return __uint_as_float(i);
}
__device__ __forceinline__ float block_sum_256(float v, float* red) {
  int t = threadIdx.x;
  red[t] = v; __syncthreads();
  #pragma unroll
  for (int s = 128; s > 0; s >>= 1) { if (t < s) red[t] += red[t + s]; __syncthreads(); }
  float r = red[0]; __syncthreads();
  return r;
}

// ---------- fills ----------
__global__ void fill_f32(float* p, float v, long long n) {
  long long i = (long long)blockIdx.x * blockDim.x + threadIdx.x;
  if (i < n) p[i] = v;
}
__global__ void fill_u32(unsigned* p, unsigned v, long long n) {
  long long i = (long long)blockIdx.x * blockDim.x + threadIdx.x;
  if (i < n) p[i] = v;
}

// ---------- encoders: Linear(K_in->256) + ReLU + LayerNorm ----------
__global__ void __launch_bounds__(256) face_encoder(
    const float* __restrict__ x, const float* __restrict__ W,
    const float* __restrict__ b, const float* __restrict__ g,
    const float* __restrict__ beta, float* __restrict__ hOut) {
  __shared__ float red[256];
  __shared__ float xr[8];
  int n = blockIdx.x, t = threadIdx.x;
  if (t < 7) xr[t] = x[(size_t)n * 7 + t];
  __syncthreads();
  float acc = b[t];
  #pragma unroll
  for (int k = 0; k < 7; k++) acc = fmaf(xr[k], W[k * HID + t], acc);
  float val = fmaxf(acc, 0.0f);
  float mu = block_sum_256(val, red) * (1.0f / 256.0f);
  float d = val - mu;
  float var = block_sum_256(d * d, red) * (1.0f / 256.0f);
  hOut[(size_t)n * HID + t] = d * rsqrtf(var + 1e-5f) * g[t] + beta[t];
}

__global__ void __launch_bounds__(256) edge_encoder_scatter(
    const float* __restrict__ xe, const float* __restrict__ W,
    const float* __restrict__ b, const float* __restrict__ g,
    const float* __restrict__ beta, const int* __restrict__ dst,
    float* __restrict__ agg, float* __restrict__ cnt) {
  __shared__ float red[256];
  __shared__ float xr[8];
  int e = blockIdx.x, t = threadIdx.x;
  if (t < 6) xr[t] = xe[(size_t)e * 6 + t];
  __syncthreads();
  float acc = b[t];
  #pragma unroll
  for (int k = 0; k < 6; k++) acc = fmaf(xr[k], W[k * HID + t], acc);
  float val = fmaxf(acc, 0.0f);
  float mu = block_sum_256(val, red) * (1.0f / 256.0f);
  float d = val - mu;
  float var = block_sum_256(d * d, red) * (1.0f / 256.0f);
  float y = d * rsqrtf(var + 1e-5f) * g[t] + beta[t];
  int dn = dst[e];
  atomicAdd(&agg[(size_t)dn * HID + t], y);
  if (t == 0) atomicAdd(&cnt[dn], 1.0f);
}

__global__ void add_mean_agg(float* __restrict__ h, const float* __restrict__ agg,
                             const float* __restrict__ cnt, long long n) {
  long long i = (long long)blockIdx.x * blockDim.x + threadIdx.x;
  if (i < n) h[i] += agg[i] / fmaxf(cnt[i >> 8], 1.0f);
}

// ---------- f32 -> f16 ----------
__global__ void to_f16(const float* __restrict__ in, _Float16* __restrict__ out, long long n) {
  long long i = (long long)blockIdx.x * blockDim.x + threadIdx.x;
  if (i < n) out[i] = (_Float16)in[i];
}

// ---------- pack W[K=256, Ncols] f32 -> f16 WMMA B operand in this code's contiguous k order ----------
// not AMD's table order (A uses the same); per 32-k chunk: lanes 0-15 hold K 0..15 of chunk, lanes 16-31 hold
// K 16..31; VGPR v packs rows {kb+2v, kb+2v+1} for column n = lane&15.
__global__ void __launch_bounds__(32) pack_w_f16(const float* __restrict__ W,
                                                 unsigned* __restrict__ out, int Ncols) {
  int tn = blockIdx.x, kc = blockIdx.y, lane = threadIdx.x;
  int ncol = tn * 16 + (lane & 15);
  int kb = (lane >> 4) * 16;
  unsigned* o = out + (((size_t)tn * 8 + kc) * 32 + lane) * 8;
  #pragma unroll
  for (int v = 0; v < 8; v++) {
    int k0 = kc * 32 + kb + 2 * v;
    union { _Float16 h[2]; unsigned u; } c;
    c.h[0] = (_Float16)W[(size_t)k0 * Ncols + ncol];
    c.h[1] = (_Float16)W[(size_t)(k0 + 1) * Ncols + ncol];
    o[v] = c.u;
  }
}

// ---------- WMMA GEMM: C[M,Ncols] = A[M,256] * Wpacked ----------
// One wave computes a 16x64 C tile (4 accumulators): A fragment loaded once per
// K-step and reused for 4 WMMAs -> 4x less A traffic than a 16x16 tiling.
__global__ void __launch_bounds__(32) gemm16x64_wmma(const _Float16* __restrict__ A,
                                                     const unsigned* __restrict__ Bp,
                                                     float* __restrict__ C,
                                                     int Mrows, int Ncols) {
  constexpr int KC = 8;  // K = 256
  int tm = blockIdx.x, tn4 = blockIdx.y, lane = threadIdx.x;
  int row = tm * 16 + (lane & 15);
  if (row >= Mrows) row = Mrows - 1;
  const _Float16* a_row = A + (size_t)row * HID;
  int kbA = (lane >> 4) * 16;  // A: lanes 0-15 -> K 0..15; lanes 16-31 -> K 16..31
  // packed B tile stride: one 16-col tile = KC*32*8 uints = 2048
  const unsigned* bp = Bp + ((size_t)(tn4 * 4) * KC * 32 + lane) * 8;
  v8f acc0 = {0.f, 0.f, 0.f, 0.f, 0.f, 0.f, 0.f, 0.f};
  v8f acc1 = acc0, acc2 = acc0, acc3 = acc0;
  #pragma unroll
  for (int kc = 0; kc < KC; kc++) {
    union { v16h v; unsigned u[8]; } a;
    const uint4* pa0 = (const uint4*)(a_row + kc * 32 + kbA);
    const uint4* pa1 = (const uint4*)(a_row + kc * 32 + 8 + kbA);
    uint4 qa0 = *pa0, qa1 = *pa1;
    a.u[0] = qa0.x; a.u[1] = qa0.y; a.u[2] = qa0.z; a.u[3] = qa0.w;
    a.u[4] = qa1.x; a.u[5] = qa1.y; a.u[6] = qa1.z; a.u[7] = qa1.w;
    #pragma unroll
    for (int j = 0; j < 4; j++) {
      union { v16h v; unsigned u[8]; } b;
      const uint4* pb = (const uint4*)(bp + (size_t)j * (KC * 32 * 8) + (size_t)kc * 256);
      uint4 qb0 = pb[0], qb1 = pb[1];
      b.u[0] = qb0.x; b.u[1] = qb0.y; b.u[2] = qb0.z; b.u[3] = qb0.w;
      b.u[4] = qb1.x; b.u[5] = qb1.y; b.u[6] = qb1.z; b.u[7] = qb1.w;
      if (j == 0)
        acc0 = __builtin_amdgcn_wmma_f32_16x16x32_f16(false, a.v, false, b.v,
                                                      (short)0, acc0, false, false);
      else if (j == 1)
        acc1 = __builtin_amdgcn_wmma_f32_16x16x32_f16(false, a.v, false, b.v,
                                                      (short)0, acc1, false, false);
      else if (j == 2)
        acc2 = __builtin_amdgcn_wmma_f32_16x16x32_f16(false, a.v, false, b.v,
                                                      (short)0, acc2, false, false);
      else
        acc3 = __builtin_amdgcn_wmma_f32_16x16x32_f16(false, a.v, false, b.v,
                                                      (short)0, acc3, false, false);
    }
    if (kc + 1 < KC) __builtin_prefetch(a_row + (kc + 1) * 32 + kbA, 0, 1);
  }
  // C layout: VGPR j -> M = j (lanes 0-15) / j+8 (lanes 16-31), N = lane&15
  int mBase = tm * 16 + ((lane < 16) ? 0 : 8);
  int nOut = tn4 * 64 + (lane & 15);
  #pragma unroll
  for (int j = 0; j < 8; j++) {
    int r = mBase + j;
    if (r < Mrows) {
      C[(size_t)r * Ncols + nOut]      = acc0[j];
      C[(size_t)r * Ncols + nOut + 16] = acc1[j];
      C[(size_t)r * Ncols + nOut + 32] = acc2[j];
      C[(size_t)r * Ncols + nOut + 48] = acc3[j];
    }
  }
}

// ---------- attention logits per (node, head) ----------
__global__ void el_er_kernel(const float* __restrict__ f, const float* __restrict__ al,
                             const float* __restrict__ ar, float* __restrict__ el,
                             float* __restrict__ er, int N) {
  int idx = blockIdx.x * blockDim.x + threadIdx.x;
  if (idx >= N * NHEAD) return;
  int n = idx >> 2, hh = idx & 3;
  const float* fr = f + (size_t)n * HID + hh * HDIM;
  float sl = 0.f, sr = 0.f;
  #pragma unroll 4
  for (int d = 0; d < HDIM; d++) {
    float v = fr[d];
    sl = fmaf(v, al[hh * HDIM + d], sl);
    sr = fmaf(v, ar[hh * HDIM + d], sr);
  }
  el[idx] = sl; er[idx] = sr;
}

// ---------- edge softmax (3 passes) ----------
__global__ void edge_logits(const int* __restrict__ src, const int* __restrict__ dst,
                            const float* __restrict__ el, const float* __restrict__ er,
                            float* __restrict__ ebuf, unsigned* __restrict__ menc,
                            long long EH) {
  long long i = (long long)blockIdx.x * blockDim.x + threadIdx.x;
  if (i >= EH) return;
  long long e = i >> 2; int hh = (int)(i & 3);
  int s = src[e], d = dst[e];
  float v = el[(size_t)s * 4 + hh] + er[(size_t)d * 4 + hh];
  v = (v > 0.f) ? v : 0.2f * v;  // leaky_relu(0.2)
  ebuf[i] = v;
  atomicMax(&menc[(size_t)d * 4 + hh], fenc(v));
}
__global__ void edge_expsum(const int* __restrict__ dst, const unsigned* __restrict__ menc,
                            float* __restrict__ ebuf, float* __restrict__ sbuf, long long EH) {
  long long i = (long long)blockIdx.x * blockDim.x + threadIdx.x;
  if (i >= EH) return;
  long long e = i >> 2; int hh = (int)(i & 3);
  int d = dst[e];
  float ex = __expf(ebuf[i] - fdec(menc[(size_t)d * 4 + hh]));
  ebuf[i] = ex;
  atomicAdd(&sbuf[(size_t)d * 4 + hh], ex);
}
__global__ void __launch_bounds__(256) edge_aggregate(const int* __restrict__ src,
                                                      const int* __restrict__ dst,
                                                      const float* __restrict__ ebuf,
                                                      const float* __restrict__ sbuf,
                                                      const float* __restrict__ f,
                                                      float* __restrict__ r) {
  int e = blockIdx.x, t = threadIdx.x, hh = t >> 6;
  int s = src[e], d = dst[e];
  float a = ebuf[(size_t)e * 4 + hh] / sbuf[(size_t)d * 4 + hh];
  atomicAdd(&r[(size_t)d * HID + t], f[(size_t)s * HID + t] * a);
}

// ---------- GAT output: ReLU(r + bias) -> LN ----------
__global__ void __launch_bounds__(256) gat_finalize(const float* __restrict__ r,
                                                    const float* __restrict__ bias,
                                                    const float* __restrict__ g,
                                                    const float* __restrict__ beta,
                                                    float* __restrict__ h) {
  __shared__ float red[256];
  int n = blockIdx.x, t = threadIdx.x;
  float val = fmaxf(r[(size_t)n * HID + t] + bias[t], 0.0f);
  float mu = block_sum_256(val, red) * (1.0f / 256.0f);
  float d = val - mu;
  float var = block_sum_256(d * d, red) * (1.0f / 256.0f);
  h[(size_t)n * HID + t] = d * rsqrtf(var + 1e-5f) * g[t] + beta[t];
}

// ---------- global attention pooling ----------
__global__ void gate_score(const float* __restrict__ h, const float* __restrict__ gw,
                           const float* __restrict__ gb, const int* __restrict__ gid,
                           float* __restrict__ gate, unsigned* __restrict__ gm, int N) {
  int n = blockIdx.x * blockDim.x + threadIdx.x;
  if (n >= N) return;
  float acc = gb[0];
  #pragma unroll 4
  for (int k = 0; k < HID; k++) acc = fmaf(h[(size_t)n * HID + k], gw[k], acc);
  gate[n] = acc;
  atomicMax(&gm[gid[n]], fenc(acc));
}
__global__ void gate_exp(const int* __restrict__ gid, const unsigned* __restrict__ gm,
                         float* __restrict__ gate, float* __restrict__ gs, int N) {
  int n = blockIdx.x * blockDim.x + threadIdx.x;
  if (n >= N) return;
  float ge = __expf(gate[n] - fdec(gm[gid[n]]));
  gate[n] = ge;
  atomicAdd(&gs[gid[n]], ge);
}
__global__ void __launch_bounds__(256) gate_pool(const float* __restrict__ h,
                                                 const float* __restrict__ gate,
                                                 const float* __restrict__ gs,
                                                 const int* __restrict__ gid,
                                                 float* __restrict__ pooled) {
  int n = blockIdx.x, t = threadIdx.x;
  int g = gid[n];
  float w = gate[n] / gs[g];
  atomicAdd(&pooled[(size_t)g * HID + t], w * h[(size_t)n * HID + t]);
}

// ---------- final: ReLU(C + bias) -> LN over 512 ----------
__global__ void __launch_bounds__(256) out_ln(const float* __restrict__ C,
                                              const float* __restrict__ bias,
                                              const float* __restrict__ g,
                                              const float* __restrict__ beta,
                                              float* __restrict__ feat) {
  __shared__ float red[256];
  int b = blockIdx.x, t = threadIdx.x;
  float v0 = fmaxf(C[(size_t)b * OUTD + t] + bias[t], 0.0f);
  float v1 = fmaxf(C[(size_t)b * OUTD + 256 + t] + bias[256 + t], 0.0f);
  float mu = block_sum_256(v0 + v1, red) * (1.0f / 512.0f);
  float d0 = v0 - mu, d1 = v1 - mu;
  float var = block_sum_256(d0 * d0 + d1 * d1, red) * (1.0f / 512.0f);
  float rs = rsqrtf(var + 1e-5f);
  feat[(size_t)b * OUTD + t]       = d0 * rs * g[t] + beta[t];
  feat[(size_t)b * OUTD + 256 + t] = d1 * rs * g[256 + t] + beta[256 + t];
}

// ======================= host side =======================
static inline long long cdiv_ll(long long a, long long b) { return (a + b - 1) / b; }

extern "C" void kernel_launch(void* const* d_in, const int* in_sizes, int n_in,
                              void* d_out, int out_size, void* d_ws, size_t ws_size,
                              hipStream_t stream) {
  const float* x_face   = (const float*)d_in[0];
  const float* x_edge   = (const float*)d_in[1];
  const float* face_W   = (const float*)d_in[2];
  const float* face_b   = (const float*)d_in[3];
  const float* face_g   = (const float*)d_in[4];
  const float* face_bt  = (const float*)d_in[5];
  const float* edge_W   = (const float*)d_in[6];
  const float* edge_b   = (const float*)d_in[7];
  const float* edge_g   = (const float*)d_in[8];
  const float* edge_bt  = (const float*)d_in[9];
  const float* gW[2]    = {(const float*)d_in[10], (const float*)d_in[16]};
  const float* gal[2]   = {(const float*)d_in[11], (const float*)d_in[17]};
  const float* gar[2]   = {(const float*)d_in[12], (const float*)d_in[18]};
  const float* gbias[2] = {(const float*)d_in[13], (const float*)d_in[19]};
  const float* gg[2]    = {(const float*)d_in[14], (const float*)d_in[20]};
  const float* gbt[2]   = {(const float*)d_in[15], (const float*)d_in[21]};
  const float* gate_W   = (const float*)d_in[22];
  const float* gate_b   = (const float*)d_in[23];
  const float* out_W    = (const float*)d_in[24];
  const float* out_b    = (const float*)d_in[25];
  const float* out_g    = (const float*)d_in[26];
  const float* out_bt   = (const float*)d_in[27];
  const int*   src      = (const int*)d_in[28];
  const int*   dst      = (const int*)d_in[29];
  const int*   gid      = (const int*)d_in[30];

  const long long N = in_sizes[0] / 7;
  const long long E = in_sizes[1] / 6;
  const long long B = ((long long)out_size - N * HID) / (OUTD + HID);

  // output layout: feat[B,512] | h[N,256] | pooled[B,256]
  float* feat   = (float*)d_out;
  float* h      = feat + (size_t)B * OUTD;
  float* pooled = h + (size_t)N * HID;

  // workspace carve
  size_t off = 0;
  auto carve = [&](size_t bytes) -> char* {
    char* p = (char*)d_ws + off;
    off += (bytes + 255) & ~(size_t)255;
    return p;
  };
  float*    f_buf  = (float*)carve((size_t)N * HID * 4);
  float*    tmp    = (float*)carve((size_t)N * HID * 4);      // agg, then r
  _Float16* h16    = (_Float16*)carve((size_t)N * HID * 2);
  unsigned* wpack  = (unsigned*)carve((size_t)HID * OUTD * 2);
  float*    e_buf  = (float*)carve((size_t)E * NHEAD * 4);
  float*    el_b   = (float*)carve((size_t)N * NHEAD * 4);
  float*    er_b   = (float*)carve((size_t)N * NHEAD * 4);
  unsigned* m_enc  = (unsigned*)carve((size_t)N * NHEAD * 4);
  float*    s_buf  = (float*)carve((size_t)N * NHEAD * 4);
  float*    cnt    = (float*)carve((size_t)N * 4);
  float*    gate   = (float*)carve((size_t)N * 4);
  unsigned* gm_enc = (unsigned*)carve((size_t)B * 4);
  float*    gs     = (float*)carve((size_t)B * 4);
  (void)ws_size; (void)n_in;

  const long long NH = N * HID, NHd = N * NHEAD, EHd = E * NHEAD;
  const dim3 blk256(256), blk32(32);

  // ---- encoders + mean edge aggregation ----
  fill_f32<<<cdiv_ll(NH, 256), blk256, 0, stream>>>(tmp, 0.f, NH);
  fill_f32<<<cdiv_ll(N, 256), blk256, 0, stream>>>(cnt, 0.f, N);
  face_encoder<<<(unsigned)N, blk256, 0, stream>>>(x_face, face_W, face_b, face_g, face_bt, h);
  edge_encoder_scatter<<<(unsigned)E, blk256, 0, stream>>>(x_edge, edge_W, edge_b, edge_g,
                                                           edge_bt, dst, tmp, cnt);
  add_mean_agg<<<cdiv_ll(NH, 256), blk256, 0, stream>>>(h, tmp, cnt, NH);

  // ---- two GAT layers ----
  for (int L = 0; L < 2; L++) {
    fill_f32<<<cdiv_ll(NH, 256), blk256, 0, stream>>>(tmp, 0.f, NH);
    fill_u32<<<cdiv_ll(NHd, 256), blk256, 0, stream>>>(m_enc, ENC_NEG_INF, NHd);
    fill_f32<<<cdiv_ll(NHd, 256), blk256, 0, stream>>>(s_buf, 0.f, NHd);
    to_f16<<<cdiv_ll(NH, 256), blk256, 0, stream>>>(h, h16, NH);
    pack_w_f16<<<dim3(HID / 16, 8), blk32, 0, stream>>>(gW[L], wpack, HID);
    gemm16x64_wmma<<<dim3((unsigned)cdiv_ll(N, 16), HID / 64), blk32, 0, stream>>>(
        h16, wpack, f_buf, (int)N, HID);
    el_er_kernel<<<cdiv_ll(NHd, 256), blk256, 0, stream>>>(f_buf, gal[L], gar[L], el_b, er_b, (int)N);
    edge_logits<<<cdiv_ll(EHd, 256), blk256, 0, stream>>>(src, dst, el_b, er_b, e_buf, m_enc, EHd);
    edge_expsum<<<cdiv_ll(EHd, 256), blk256, 0, stream>>>(dst, m_enc, e_buf, s_buf, EHd);
    edge_aggregate<<<(unsigned)E, blk256, 0, stream>>>(src, dst, e_buf, s_buf, f_buf, tmp);
    gat_finalize<<<(unsigned)N, blk256, 0, stream>>>(tmp, gbias[L], gg[L], gbt[L], h);
  }

  // ---- global attention pooling ----
  fill_f32<<<cdiv_ll(B * HID, 256), blk256, 0, stream>>>(pooled, 0.f, B * HID);
  fill_u32<<<1, blk256, 0, stream>>>(gm_enc, ENC_NEG_INF, B);
  fill_f32<<<1, blk256, 0, stream>>>(gs, 0.f, B);
  gate_score<<<cdiv_ll(N, 256), blk256, 0, stream>>>(h, gate_W, gate_b, gid, gate, gm_enc, (int)N);
  gate_exp<<<cdiv_ll(N, 256), blk256, 0, stream>>>(gid, gm_enc, gate, gs, (int)N);
  gate_pool<<<(unsigned)N, blk256, 0, stream>>>(h, gate, gs, gid, pooled);

  // ---- output head: pooled[B,256] @ out_W[256,512] -> ReLU+LN ----
  fill_u32<<<cdiv_ll(16 * HID / 2, 256), blk256, 0, stream>>>((unsigned*)h16, 0u, 16 * HID / 2);
  to_f16<<<cdiv_ll(B * HID, 256), blk256, 0, stream>>>(pooled, h16, B * HID);
  pack_w_f16<<<dim3(OUTD / 16, 8), blk32, 0, stream>>>(out_W, wpack, OUTD);
  gemm16x64_wmma<<<dim3((unsigned)cdiv_ll(B, 16), OUTD / 64), blk32, 0, stream>>>(
      h16, wpack, f_buf, (int)B, OUTD);
  out_ln<<<(unsigned)B, blk256, 0, stream>>>(f_buf, out_b, out_g, out_bt, feat);
}
